// GraphSAGE_node_calssfication_sobel_77180562309272
// MI455X (gfx1250) — matrix operand order repaired and checked by reading, not compiled, not run on hardware
//
#include <hip/hip_runtime.h>
#include <hip/hip_bf16.h>

typedef __attribute__((ext_vector_type(16))) _Float16 v16h;
typedef __attribute__((ext_vector_type(8)))  float    v8f;

#define NN 50000
#define NE 800000
#define IND 128
#define HID 256
#define OUTD 5

// ---------------- degree count (float so the reciprocal pass is trivial) ----------------
__global__ void count_edges(const long long* __restrict__ dst, float* __restrict__ cnt, int ne) {
    int i = blockIdx.x * blockDim.x + threadIdx.x;
    if (i < ne) atomicAdd(&cnt[(int)dst[i]], 1.0f);
}

__global__ void make_inv(const float* __restrict__ cnt, float* __restrict__ inv, int n) {
    int i = blockIdx.x * blockDim.x + threadIdx.x;
    if (i < n) inv[i] = 1.0f / fmaxf(cnt[i], 1.0f);
}

// ---------------- edge-parallel scatter-add: agg[dst] += X[src] (L2-resident atomics) ----
template <int DIM>
__global__ void scatter_add(const long long* __restrict__ src, const long long* __restrict__ dst,
                            const float* __restrict__ X, float* __restrict__ Agg, int ne) {
    constexpr int PER_EDGE = DIM / 4;                     // float4 per thread
    long long i = (long long)blockIdx.x * blockDim.x + threadIdx.x;
    long long total = (long long)ne * PER_EDGE;
    if (i >= total) return;
    int e = (int)(i / PER_EDGE);
    int c = (int)(i % PER_EDGE) << 2;                     // feature offset
    int s = (int)src[e], d = (int)dst[e];
    const float4 v = *reinterpret_cast<const float4*>(X + (size_t)s * DIM + c);
    float* o = Agg + (size_t)d * DIM + c;
    atomicAdd(o + 0, v.x); atomicAdd(o + 1, v.y);
    atomicAdd(o + 2, v.z); atomicAdd(o + 3, v.w);
}

// ---------------- pre-convert weights: Wc = [Wl | Wr] along K, f32 -> f16 ----------------
__global__ void prep_weights(const float* __restrict__ W1l, const float* __restrict__ W1r,
                             const float* __restrict__ W2l, const float* __restrict__ W2r,
                             _Float16* __restrict__ W1h,    // [256][256]
                             _Float16* __restrict__ W2h) {  // [16][512], rows 5..15 zero
    int i = blockIdx.x * blockDim.x + threadIdx.x;
    if (i < 256 * 256) {
        int o = i >> 8, k = i & 255;
        float v = (k < 128) ? W1l[o * 128 + k] : W1r[o * 128 + (k - 128)];
        W1h[i] = (_Float16)v;
    } else {
        int j = i - 256 * 256;
        if (j < 16 * 512) {
            int o = j >> 9, k = j & 511;
            float v = 0.0f;
            if (o < 5) v = (k < 256) ? W2l[o * 256 + k] : W2r[o * 256 + (k - 256)];
            W2h[j] = (_Float16)v;
        }
    }
}

// ---------------- fused SAGE layer GEMM via WMMA -----------------------------------------
// Out[i, o] = relu( bias[o] + sum_k [inv[i]*Aagg[i,:] | Aself[i,:]][k] * Wh[o][k] )
// One wave computes one 16x16 tile; K = 2*KHALF, stepped 32 per v_wmma_f32_16x16x32_f16.
template <int KHALF>
__global__ void sage_gemm_wmma(const float* __restrict__ Aagg,   // [rows, KHALF]
                               const float* __restrict__ Aself,  // [rows, KHALF]
                               const float* __restrict__ inv,    // [rows]
                               const _Float16* __restrict__ Wh,  // [ntile_n*16, 2*KHALF]
                               const float* __restrict__ bias,
                               float* __restrict__ Out,
                               int ntile_m, int ntile_n, int ncols, int outstride) {
    constexpr int K  = 2 * KHALF;
    constexpr int KT = K / 32;
    int wave  = threadIdx.x >> 5;
    int tile  = blockIdx.x * (blockDim.x >> 5) + wave;
    if (tile >= ntile_m * ntile_n) return;
    int tm = tile / ntile_n, tn = tile % ntile_n;
    int lane  = threadIdx.x & 31;
    int half_ = lane >> 4;          // K-half selector (A/B) / M-half (C,D)
    int lq    = lane & 15;          // M for A, N for B/C/D

    int arow = tm * 16 + lq;
    float sc_agg = inv[arow];
    const float*    aggrow  = Aagg  + (size_t)arow * KHALF;
    const float*    selfrow = Aself + (size_t)arow * KHALF;
    const _Float16* wrow    = Wh + (size_t)(tn * 16 + lq) * K;

    v8f acc = {};
#pragma unroll
    for (int kt = 0; kt < KT; ++kt) {
        int kbase = kt * 32;
        // ---- A fragment: lane holds row M=lq,
        //      elements t=0..15 -> K = kbase + 16*half_ + t (matches B).
        const float* ap; float sc;
        if (kbase < KHALF) { ap = aggrow  + kbase;         sc = sc_agg; }
        else               { ap = selfrow + (kbase-KHALF); sc = 1.0f;  }
        float4 f0 = *reinterpret_cast<const float4*>(ap + 16 * half_);
        float4 f1 = *reinterpret_cast<const float4*>(ap + 16 * half_ + 4);
        float4 f2 = *reinterpret_cast<const float4*>(ap + 16 * half_ + 8);
        float4 f3 = *reinterpret_cast<const float4*>(ap + 16 * half_ + 12);
        v16h a;
        a[0]=(_Float16)(f0.x*sc); a[1]=(_Float16)(f0.y*sc); a[2]=(_Float16)(f0.z*sc); a[3]=(_Float16)(f0.w*sc);
        a[4]=(_Float16)(f1.x*sc); a[5]=(_Float16)(f1.y*sc); a[6]=(_Float16)(f1.z*sc); a[7]=(_Float16)(f1.w*sc);
        a[8]=(_Float16)(f2.x*sc); a[9]=(_Float16)(f2.y*sc); a[10]=(_Float16)(f2.z*sc); a[11]=(_Float16)(f2.w*sc);
        a[12]=(_Float16)(f3.x*sc); a[13]=(_Float16)(f3.y*sc); a[14]=(_Float16)(f3.z*sc); a[15]=(_Float16)(f3.w*sc);
        // ---- B fragment: lane holds column N=lq, 16 contiguous halfs at K = kbase+16*half_.
        v16h b = *reinterpret_cast<const v16h*>(wrow + kbase + 16 * half_);
        acc = __builtin_amdgcn_wmma_f32_16x16x32_f16(false, a, false, b, (short)0, acc, false, false);
    }

    int o = tn * 16 + lq;                       // output channel = N
    if (o < ncols) {
        float bv = bias[o];
#pragma unroll
        for (int r = 0; r < 8; ++r) {           // acc[r] -> row M = r + 8*half_
            int row = tm * 16 + 8 * half_ + r;
            float v = acc[r] + bv;
            Out[(size_t)row * outstride + o] = v > 0.0f ? v : 0.0f;
        }
    }
}

// ---------------- launch ------------------------------------------------------------------
extern "C" void kernel_launch(void* const* d_in, const int* in_sizes, int n_in,
                              void* d_out, int out_size, void* d_ws, size_t ws_size,
                              hipStream_t stream) {
    (void)in_sizes; (void)n_in; (void)out_size; (void)ws_size;
    const float*     x    = (const float*)d_in[0];
    const long long* ei   = (const long long*)d_in[1];   // int64 [2, NE]
    const float*     W1l  = (const float*)d_in[2];
    const float*     b1l  = (const float*)d_in[3];
    const float*     W1r  = (const float*)d_in[4];
    const float*     W2l  = (const float*)d_in[5];
    const float*     b2l  = (const float*)d_in[6];
    const float*     W2r  = (const float*)d_in[7];
    float*           out  = (float*)d_out;

    // Workspace carve (256B aligned). cnt|agg1|agg2 kept adjacent for one memset.
    char* p = (char*)d_ws;
    auto carve = [&](size_t bytes) { char* r = p; p += (bytes + 255) & ~(size_t)255; return r; };
    float*    cnt  = (float*)   carve((size_t)NN * 4);
    float*    agg1 = (float*)   carve((size_t)NN * IND * 4);
    float*    agg2 = (float*)   carve((size_t)NN * HID * 4);
    float*    inv  = (float*)   carve((size_t)NN * 4);
    float*    h    = (float*)   carve((size_t)NN * HID * 4);
    _Float16* W1h  = (_Float16*)carve((size_t)256 * 256 * 2);
    _Float16* W2h  = (_Float16*)carve((size_t)16 * 512 * 2);

    const long long* src = ei;            // row 0
    const long long* dst = ei + NE;       // row 1

    // Zero counters + accumulators (single capturable memset over contiguous region).
    hipMemsetAsync(cnt, 0, (size_t)((char*)inv - (char*)cnt), stream);

    prep_weights<<<(256 * 256 + 16 * 512 + 255) / 256, 256, 0, stream>>>(W1l, W1r, W2l, W2r, W1h, W2h);
    count_edges<<<(NE + 255) / 256, 256, 0, stream>>>(dst, cnt, NE);
    make_inv<<<(NN + 255) / 256, 256, 0, stream>>>(cnt, inv, NN);

    // Layer 1: aggregate x, then fused [mean|x] @ [W1l|W1r]^T + b1l, ReLU -> h
    {
        long long tot = (long long)NE * (IND / 4);
        scatter_add<IND><<<(int)((tot + 255) / 256), 256, 0, stream>>>(src, dst, x, agg1, NE);
    }
    sage_gemm_wmma<128><<<(3125 * 16) / 8, 256, 0, stream>>>(agg1, x, inv, W1h, b1l, h,
                                                             3125, 16, HID, HID);

    // Layer 2: aggregate h, then fused [mean|h] @ [W2l|W2r]^T + b2l, ReLU -> out [50000,5]
    {
        long long tot = (long long)NE * (HID / 4);
        scatter_add<HID><<<(int)((tot + 255) / 256), 256, 0, stream>>>(src, dst, h, agg2, NE);
    }
    sage_gemm_wmma<256><<<(3125 + 7) / 8, 256, 0, stream>>>(agg2, h, inv, W2h, b2l, out,
                                                            3125, 1, OUTD, OUTD);
}
